// GMS_14147622273713
// MI455X (gfx1250) — matrix operand order repaired and checked by reading, not compiled, not run on hardware
//
#include <hip/hip_runtime.h>

#define EMBD 128

typedef __attribute__((ext_vector_type(16))) __bf16 v16bf;
typedef __attribute__((ext_vector_type(8)))  float  v8f;

__device__ __forceinline__ unsigned short f2bf(float f) {
  unsigned int u = __float_as_uint(f);
  u += 0x7FFFu + ((u >> 16) & 1u);           // round to nearest even
  return (unsigned short)(u >> 16);
}

__device__ __forceinline__ float sigf(float x) {
  return 1.0f / (1.0f + __expf(-x));
}

// ---------------------------------------------------------------------------
// Weight packer: logical B[k][n] (K=128, N columns) -> WMMA-fragment-major bf16.
//   out[((kt*nN + nt)*32 + lane)*16 + e] = bf16( B[kt*32 + 16*(e>>3) + 8*(lane>>4) + (e&7)][nt*16 + (lane&15)] )
// (B-fragment lane map: lanes 0-15 hold K=k0..k0+7,k0+16..k0+23; lanes 16-31 hold K=k0+8..k0+15,k0+24..k0+31.)
// transpose=0: B[k][n] = w[k*N + n]    (w stored [128][N])
// transpose=1: B[k][n] = w[n*128 + k]  (w stored [N][128], torch LSTM weights)
// ---------------------------------------------------------------------------
__global__ void pack_b_k(const float* __restrict__ w, unsigned short* __restrict__ out,
                         int N, int transpose) {
  const int i = blockIdx.x * 256 + threadIdx.x;
  if (i >= 128 * N) return;
  const int e    = i & 15;
  const int lane = (i >> 4) & 31;
  const int rest = i >> 9;
  const int nN   = N >> 4;
  const int nt   = rest % nN;
  const int kt   = rest / nN;
  const int k = kt * 32 + ((e >> 3) << 4) + ((lane >> 4) << 3) + (e & 7);
  const int n = nt * 16 + (lane & 15);
  const float v = transpose ? w[n * 128 + k] : w[k * N + n];
  out[i] = f2bf(v);
}

__global__ void vec_add_k(const float* __restrict__ a, const float* __restrict__ b,
                          float* __restrict__ o, int n) {
  int i = blockIdx.x * 256 + threadIdx.x;
  if (i < n) o[i] = a[i] + b[i];
}
__global__ void copy_f_k(const float* __restrict__ s, float* __restrict__ d, int n) {
  int i = blockIdx.x * 256 + threadIdx.x;
  if (i < n) d[i] = s[i];
}
__global__ void zero_f_k(float* __restrict__ d, int n) {
  int i = blockIdx.x * 256 + threadIdx.x;
  if (i < n) d[i] = 0.0f;
}

// A-fragment packed LDS staging helper: a float4 of X at (row m, k=c4, c4%4==0)
// maps to 4 contiguous e-slots of lane (h*16+m) in k-tile kt:
//   kt = c4>>5, kk = c4&31, h = (kk>>3)&1, e0 = ((kk&16)>>1) + (kk&7)
__device__ __forceinline__ void stage_a4(unsigned short* xs, int frag_base32,
                                         int m, int c4, float4 xv) {
  const int kt = c4 >> 5, kk = c4 & 31;
  const int h  = (kk >> 3) & 1;
  const int e0 = ((kk & 16) >> 1) + (kk & 7);
  unsigned short* p = &xs[(((frag_base32 + kt) << 5) + (h << 4) + m) * 16 + e0];
  p[0] = f2bf(xv.x); p[1] = f2bf(xv.y); p[2] = f2bf(xv.z); p[3] = f2bf(xv.w);
}

// ---------------- MLP layer: Y = act(X @ W + b), 128x128 weights ----------------
// Block: 128 rows, 256 threads / 8 waves. Wave w owns column stripe [16w,16w+16):
// its 4 B fragments live in registers and are reused across 8 row tiles.
template <bool RELU>
__global__ void mlp_layer_k(const float* __restrict__ X, const unsigned short* __restrict__ Wp,
                            const float* __restrict__ bias, float* __restrict__ Y, int nrows) {
  __shared__ __align__(32) unsigned short xs[8 * 4 * 32 * 16]; // [rt][kt][lane][e], 32KB
  const int tid = threadIdx.x, lane = tid & 31, wave = tid >> 5;
  const int rowBase = blockIdx.x * 128;

  for (int i = tid; i < 128 * 32; i += 256) {
    const int r = i >> 5, c4 = (i & 31) << 2;
    const int gr = rowBase + r;
    float4 xv = make_float4(0.f, 0.f, 0.f, 0.f);
    if (gr < nrows) xv = *(const float4*)(X + (size_t)gr * EMBD + c4);
    stage_a4(xs, (r >> 4) << 2, r & 15, c4, xv);
  }
  __syncthreads();

  const int n = lane & 15, hh = lane >> 4;
  const int col0 = wave * 16;
  const v16bf* wp = (const v16bf*)Wp;
  v16bf bf[4];
#pragma unroll
  for (int kt = 0; kt < 4; ++kt)
    bf[kt] = wp[(kt * 8 + wave) * 32 + lane];      // 2x global_load_b128 each
  const float bv = bias[col0 + n];
  const v16bf* ap = (const v16bf*)xs;

#pragma unroll
  for (int rt = 0; rt < 8; ++rt) {
    v8f acc;
#pragma unroll
    for (int v = 0; v < 8; ++v) acc[v] = bv;
#pragma unroll
    for (int kt = 0; kt < 4; ++kt) {
      v16bf a = ap[(rt * 4 + kt) * 32 + lane];     // 2x ds_load_b128
      acc = __builtin_amdgcn_wmma_f32_16x16x32_bf16(false, a, false, bf[kt],
                                                    (short)0, acc, false, false);
    }
#pragma unroll
    for (int v = 0; v < 8; ++v) {
      const int gr = rowBase + rt * 16 + v + 8 * hh;
      if (gr < nrows) {
        float o = acc[v];
        if (RELU) o = o > 0.f ? o : 0.f;
        Y[(size_t)gr * EMBD + col0 + n] = o;
      }
    }
  }
}

// ---------------- edge scatter: out[sidx[e]] += msg[gidx[e]] (rowwise) ----------------
__global__ void scatter_add_k(const float* __restrict__ msg, const int* __restrict__ gidx,
                              const int* __restrict__ sidx, float* __restrict__ out, int ne) {
  const int t = blockIdx.x * 256 + threadIdx.x;
  if (t >= ne * 32) return;
  const int e = t >> 5;
  const int c4 = (t & 31) << 2;
  const int g = gidx[e], s = sidx[e];
  const float4 v = *(const float4*)(msg + (size_t)g * EMBD + c4);
  float* o = out + (size_t)s * EMBD + c4;
  atomicAdd(o + 0, v.x); atomicAdd(o + 1, v.y);
  atomicAdd(o + 2, v.z); atomicAdd(o + 3, v.w);
}

// ---------------- fused LSTM cell ----------------
// 16-row tile per block. gates = x@Bih + h@Bhh + bsum (row-wise -> per-block safe).
// x/h A-fragments hoisted to registers; gates staged in LDS; pointwise update in-place.
__global__ void lstm_fused_k(const float* __restrict__ Xin, int xmod,
                             float* __restrict__ H, float* __restrict__ C,
                             const unsigned short* __restrict__ Bih,
                             const unsigned short* __restrict__ Bhh,
                             const float* __restrict__ bsum, int nrows) {
  __shared__ __align__(32) unsigned short xsh[4 * 32 * 16]; // 4KB
  __shared__ __align__(32) unsigned short hsh[4 * 32 * 16]; // 4KB
  __shared__ float gbuf[16 * 512];                          // 32KB
  const int tid = threadIdx.x, lane = tid & 31, wave = tid >> 5;
  const int rowBase = blockIdx.x * 16;

  for (int i = tid; i < 16 * 32; i += 256) {
    const int r = i >> 5, c4 = (i & 31) << 2;
    const int gr = rowBase + r;
    float4 xv = make_float4(0.f, 0.f, 0.f, 0.f), hv = xv;
    if (gr < nrows) {
      const int xr = xmod ? (gr % xmod) : gr;
      xv = *(const float4*)(Xin + (size_t)xr * EMBD + c4);
      hv = *(const float4*)(H + (size_t)gr * EMBD + c4);
    }
    stage_a4(xsh, 0, r, c4, xv);
    stage_a4(hsh, 0, r, c4, hv);
  }
  __syncthreads();

  const int n = lane & 15, hh = lane >> 4;
  const v16bf* apx = (const v16bf*)xsh;
  const v16bf* aph = (const v16bf*)hsh;
  v16bf xa[4], ha[4];
#pragma unroll
  for (int kt = 0; kt < 4; ++kt) {
    xa[kt] = apx[kt * 32 + lane];
    ha[kt] = aph[kt * 32 + lane];
  }
  const v16bf* bih = (const v16bf*)Bih;
  const v16bf* bhh = (const v16bf*)Bhh;

#pragma unroll
  for (int nt = 0; nt < 4; ++nt) {
    const int ntAbs = wave * 4 + nt;        // 0..31 over the 512 gate columns
    const int col0 = ntAbs * 16;
    v8f acc;
    const float bv = bsum[col0 + n];
#pragma unroll
    for (int v = 0; v < 8; ++v) acc[v] = bv;
#pragma unroll
    for (int kt = 0; kt < 4; ++kt) {
      v16bf b = bih[(kt * 32 + ntAbs) * 32 + lane];
      acc = __builtin_amdgcn_wmma_f32_16x16x32_bf16(false, xa[kt], false, b,
                                                    (short)0, acc, false, false);
    }
#pragma unroll
    for (int kt = 0; kt < 4; ++kt) {
      v16bf b = bhh[(kt * 32 + ntAbs) * 32 + lane];
      acc = __builtin_amdgcn_wmma_f32_16x16x32_bf16(false, ha[kt], false, b,
                                                    (short)0, acc, false, false);
    }
#pragma unroll
    for (int v = 0; v < 8; ++v)
      gbuf[(v + 8 * hh) * 512 + col0 + n] = acc[v];
  }
  __syncthreads();

  for (int i = tid; i < 16 * 128; i += 256) {
    const int r = i >> 7, d = i & 127;
    const int gr = rowBase + r;
    if (gr < nrows) {
      const float gi = gbuf[r * 512 + d];
      const float gf = gbuf[r * 512 + 128 + d];
      const float gg = gbuf[r * 512 + 256 + d];
      const float go = gbuf[r * 512 + 384 + d];
      const size_t off = (size_t)gr * EMBD + d;
      const float cn = sigf(gf) * C[off] + sigf(gi) * tanhf(gg);
      C[off] = cn;
      H[off] = sigf(go) * tanhf(cn);
    }
  }
}

// ---------------- host orchestration ----------------
static inline int cdiv(int a, int b) { return (a + b - 1) / b; }

static void run_mlp3(const float* X, int nrows,
                     const unsigned short* const* w, const float* const* b,
                     float* tmpA, float* tmpB, float* out, hipStream_t s) {
  const int blocks = cdiv(nrows, 128);
  mlp_layer_k<true ><<<blocks, 256, 0, s>>>(X,    w[0], b[0], tmpA, nrows);
  mlp_layer_k<true ><<<blocks, 256, 0, s>>>(tmpA, w[1], b[1], tmpB, nrows);
  mlp_layer_k<false><<<blocks, 256, 0, s>>>(tmpB, w[2], b[2], out,  nrows);
}

extern "C" void kernel_launch(void* const* d_in, const int* in_sizes, int n_in,
                              void* d_out, int out_size, void* d_ws, size_t ws_size,
                              hipStream_t stream) {
  (void)n_in; (void)out_size; (void)ws_size;
  const int NV = in_sizes[0] / EMBD;
  const int NC = in_sizes[2] / EMBD;
  const int NE = in_sizes[3];
  const int NUM_ROUND = 16;

  const float* lp = (const float*)d_in[0];
  const float* ln = (const float*)d_in[1];
  const float* ce = (const float*)d_in[2];
  const int* pos_src = (const int*)d_in[3];
  const int* pos_dst = (const int*)d_in[4];
  const int* neg_src = (const int*)d_in[5];
  const int* neg_dst = (const int*)d_in[6];
  // params flattened: sorted dict keys -> c_update{b_hh,b_ih,w_hh,w_ih},
  // l_update{...}, neg_c_msg[w0,b0,w1,b1,w2,b2], neg_l_msg, pos_c_msg, pos_l_msg
  const float* c_b_hh = (const float*)d_in[7];
  const float* c_b_ih = (const float*)d_in[8];
  const float* c_w_hh = (const float*)d_in[9];
  const float* c_w_ih = (const float*)d_in[10];
  const float* l_b_hh = (const float*)d_in[11];
  const float* l_b_ih = (const float*)d_in[12];
  const float* l_w_hh = (const float*)d_in[13];
  const float* l_w_ih = (const float*)d_in[14];
  // mlp order here: 0=pos_l, 1=neg_l, 2=pos_c, 3=neg_c
  const int base[4] = {33, 21, 27, 15};
  const float* mlp_wf[4][3]; const float* mlp_b[4][3];
  for (int m = 0; m < 4; ++m)
    for (int l = 0; l < 3; ++l) {
      mlp_wf[m][l] = (const float*)d_in[base[m] + 2 * l];
      mlp_b [m][l] = (const float*)d_in[base[m] + 2 * l + 1];
    }

  // states: l_h/c_h live directly in d_out (its layout IS the output)
  float* l_h = (float*)d_out;
  float* c_h = l_h + (size_t)2 * NV * EMBD;

  float* ws = (float*)d_ws;
  size_t o = 0;
  float* l_c  = ws + o; o += (size_t)2 * NV * EMBD;
  float* c_c  = ws + o; o += (size_t)NC * EMBD;
  float* bufA = ws + o; o += (size_t)NC * EMBD;   // mlp tmp0 / c->l sum target
  float* bufB = ws + o; o += (size_t)NC * EMBD;   // mlp tmp1
  float* bufC = ws + o; o += (size_t)NC * EMBD;   // pos messages
  float* bufD = ws + o; o += (size_t)NC * EMBD;   // neg messages
  float* bufE = ws + o; o += (size_t)NC * EMBD;   // l2c
  float* bsum_l = ws + o; o += 512;
  float* bsum_c = ws + o; o += 512;
  unsigned short* wb = (unsigned short*)(ws + o);
  const unsigned short* mw[4][3];
  for (int m = 0; m < 4; ++m)
    for (int l = 0; l < 3; ++l) mw[m][l] = wb + (size_t)(m * 3 + l) * 16384;
  unsigned short* Bih_l = wb + (size_t)12 * 16384;
  unsigned short* Bhh_l = Bih_l + 65536;
  unsigned short* Bih_c = Bhh_l + 65536;
  unsigned short* Bhh_c = Bih_c + 65536;

  // ---- per-launch prep (deterministic) ----
  for (int m = 0; m < 4; ++m)
    for (int l = 0; l < 3; ++l)
      pack_b_k<<<cdiv(16384, 256), 256, 0, stream>>>(mlp_wf[m][l],
                                                     (unsigned short*)mw[m][l], 128, 0);
  pack_b_k<<<cdiv(65536, 256), 256, 0, stream>>>(l_w_ih, Bih_l, 512, 1);
  pack_b_k<<<cdiv(65536, 256), 256, 0, stream>>>(l_w_hh, Bhh_l, 512, 1);
  pack_b_k<<<cdiv(65536, 256), 256, 0, stream>>>(c_w_ih, Bih_c, 512, 1);
  pack_b_k<<<cdiv(65536, 256), 256, 0, stream>>>(c_w_hh, Bhh_c, 512, 1);
  vec_add_k<<<2, 256, 0, stream>>>(l_b_ih, l_b_hh, bsum_l, 512);
  vec_add_k<<<2, 256, 0, stream>>>(c_b_ih, c_b_hh, bsum_c, 512);

  const int nvE = NV * EMBD, ncE = NC * EMBD;
  copy_f_k<<<cdiv(nvE, 256), 256, 0, stream>>>(lp, l_h, nvE);
  copy_f_k<<<cdiv(nvE, 256), 256, 0, stream>>>(ln, l_h + (size_t)nvE, nvE);
  copy_f_k<<<cdiv(ncE, 256), 256, 0, stream>>>(ce, c_h, ncE);
  zero_f_k<<<cdiv(2 * nvE, 256), 256, 0, stream>>>(l_c, 2 * nvE);
  zero_f_k<<<cdiv(ncE, 256), 256, 0, stream>>>(c_c, ncE);

  const int sblocks = cdiv(NE * 32, 256);
  for (int round = 0; round < NUM_ROUND; ++round) {
    // literal -> clause messages
    run_mlp3(l_h,               NV, mw[0], mlp_b[0], bufA, bufB, bufC, stream);
    run_mlp3(l_h + (size_t)nvE, NV, mw[1], mlp_b[1], bufA, bufB, bufD, stream);
    zero_f_k<<<cdiv(ncE, 256), 256, 0, stream>>>(bufE, ncE);
    scatter_add_k<<<sblocks, 256, 0, stream>>>(bufC, pos_src, pos_dst, bufE, NE);
    scatter_add_k<<<sblocks, 256, 0, stream>>>(bufD, neg_src, neg_dst, bufE, NE);
    // clause -> literal messages; c2l+flip == broadcast of (pos_c2l + neg_c2l)
    run_mlp3(c_h, NC, mw[2], mlp_b[2], bufA, bufB, bufC, stream);
    run_mlp3(c_h, NC, mw[3], mlp_b[3], bufA, bufB, bufD, stream);
    zero_f_k<<<cdiv(nvE, 256), 256, 0, stream>>>(bufA, nvE);
    scatter_add_k<<<sblocks, 256, 0, stream>>>(bufC, pos_dst, pos_src, bufA, NE);
    scatter_add_k<<<sblocks, 256, 0, stream>>>(bufD, neg_dst, neg_src, bufA, NE);
    // LSTM updates
    lstm_fused_k<<<cdiv(2 * NV, 16), 256, 0, stream>>>(bufA, NV, l_h, l_c,
                                                       Bih_l, Bhh_l, bsum_l, 2 * NV);
    lstm_fused_k<<<cdiv(NC, 16), 256, 0, stream>>>(bufE, 0, c_h, c_c,
                                                   Bih_c, Bhh_c, bsum_c, NC);
  }
}
